// GraphClassifier_22213570855407
// MI455X (gfx1250) — matrix operand order repaired and checked by reading, not compiled, not run on hardware
//
#include <hip/hip_runtime.h>

#define GN1 4096
#define GN2 4096

typedef __bf16 bf16_t;
typedef __attribute__((ext_vector_type(16))) __bf16 v16bf;
typedef __attribute__((ext_vector_type(8)))  __bf16 v8bf;
typedef __attribute__((ext_vector_type(8)))  float  v8f;

__device__ __forceinline__ unsigned short f32_to_bf16_bits(float f) {
  union { float f; unsigned int u; } c; c.f = f;
  unsigned int u = c.u;
  u += 0x7FFFu + ((u >> 16) & 1u);           // round to nearest even
  return (unsigned short)(u >> 16);
}

// ---------------------------------------------------------------- converts
__global__ __launch_bounds__(256)
void k_f32_to_bf16(const float* __restrict__ src, unsigned short* __restrict__ dst,
                   size_t n) {
  size_t stride = (size_t)gridDim.x * blockDim.x;
  for (size_t i = (size_t)blockIdx.x * blockDim.x + threadIdx.x; i < n; i += stride)
    dst[i] = f32_to_bf16_bits(src[i]);
}

// mask = (adj==1) * alpha  ->  bf16 ;  deg[row] = rowsum(adj==1) as f32
__global__ __launch_bounds__(256)
void k_mask_alpha(const int* __restrict__ adj, const float* __restrict__ alpha,
                  unsigned short* __restrict__ out, float* __restrict__ deg, int N) {
  __shared__ float sred[256];
  int row = blockIdx.x;
  const int*   ar = adj   + (size_t)row * N;
  const float* al = alpha + (size_t)row * N;
  unsigned short* orow = out + (size_t)row * N;
  float cnt = 0.f;
  for (int c = threadIdx.x; c < N; c += 256) {
    int m = (ar[c] == 1) ? 1 : 0;
    cnt += (float)m;
    orow[c] = f32_to_bf16_bits(m ? al[c] : 0.0f);
  }
  sred[threadIdx.x] = cnt;
  __syncthreads();
  for (int s = 128; s > 0; s >>= 1) {
    if (threadIdx.x < s) sred[threadIdx.x] += sred[threadIdx.x + s];
    __syncthreads();
  }
  if (threadIdx.x == 0) deg[row] = sred[0];
}

// ---------------------------------------------------------------- WMMA GEMM
// C[M,N] = A[M,K](bf16,row-major) @ B[N,K](bf16,row-major)^T, f32 accumulate.
// MODE 0: +bias, store bf16 C         (fc1, fc2)
// MODE 1: +bias, relu, store f32 H[M,64] and bf16 HT[64,M]   (fc3)
// MODE 2: v*W00/max(deg,1) + Hres, zero where deg==0, store f32 (attention)
// Block = 128 threads (4 waves). Wave tile = (16*MT) x 64. Explicit
// double-buffered software pipeline: fragments for iteration i+1 are loaded
// before the WMMAs of iteration i so loads stay outstanding across the MMAs.
// Requires K % 64 == 0 (true for 4096/256/128).
template <int MODE, int MT>
__global__ __launch_bounds__(128)
void gemm_tn(const unsigned short* __restrict__ A,
             const unsigned short* __restrict__ B,
             int M, int N, int K,
             const float* __restrict__ bias,
             unsigned short* __restrict__ Cb,
             float* __restrict__ Hf,
             unsigned short* __restrict__ HT,
             const float* __restrict__ Wp,
             const float* __restrict__ deg,
             const float* __restrict__ Hres,
             float* __restrict__ Of) {
  const int lane = threadIdx.x & 31;
  const int wave = threadIdx.x >> 5;
  const int li = lane & 15;   // row (A) / col (B) / col (D) within tile
  const int lh = lane >> 4;   // K-half select (A,B) / M-half select (D)
  const int m0 = blockIdx.y * (64 * MT) + wave * (16 * MT);
  const int n0 = blockIdx.x * 64;

  const bf16_t* Ab = (const bf16_t*)A;
  const bf16_t* Bb = (const bf16_t*)B;
  // A lane layout (16x32 bf16): lanes 0-15 row=li K=0..15;
  //                             lanes 16-31 row=li K=16..31
  const bf16_t* aRow[MT];
#pragma unroll
  for (int mt = 0; mt < MT; ++mt)
    aRow[mt] = Ab + (size_t)(m0 + mt * 16 + li) * K + (size_t)lh * 16;
  // B lane layout (32x16 bf16): lanes 0-15 col=li K=0..15; lanes 16-31 K=16..31
  const bf16_t* bRow = Bb + (size_t)(n0 + li) * K + (size_t)lh * 16;
  const size_t bstep = (size_t)16 * K;

  v8f acc[MT][4];
#pragma unroll
  for (int mt = 0; mt < MT; ++mt)
#pragma unroll
    for (int nt = 0; nt < 4; ++nt) acc[mt][nt] = (v8f){};

  auto load_frags = [&](int k0, v16bf af[MT], v16bf bf[4]) {
#pragma unroll
    for (int mt = 0; mt < MT; ++mt) {
      v8bf alo = *(const v8bf*)(aRow[mt] + k0);
      v8bf ahi = *(const v8bf*)(aRow[mt] + k0 + 8);
#pragma unroll
      for (int i = 0; i < 8; ++i) { af[mt][i] = alo[i]; af[mt][i + 8] = ahi[i]; }
    }
#pragma unroll
    for (int nt = 0; nt < 4; ++nt)
      bf[nt] = *(const v16bf*)(bRow + (size_t)nt * bstep + k0);
  };
  auto mma = [&](const v16bf af[MT], const v16bf bf[4]) {
#pragma unroll
    for (int nt = 0; nt < 4; ++nt)
#pragma unroll
      for (int mt = 0; mt < MT; ++mt)
        acc[mt][nt] = __builtin_amdgcn_wmma_f32_16x16x32_bf16(
            false, af[mt], false, bf[nt], (short)0, acc[mt][nt], false, false);
  };

  v16bf af0[MT], bf0[4], af1[MT], bf1[4];
  load_frags(0, af0, bf0);
  int k0 = 0;
  for (; k0 < K - 64; k0 += 64) {
    load_frags(k0 + 32, af1, bf1);   // prefetch half-step i+1
    mma(af0, bf0);
    load_frags(k0 + 64, af0, bf0);   // prefetch half-step i+2
    mma(af1, bf1);
  }
  load_frags(k0 + 32, af1, bf1);     // tail (k0 == K-64)
  mma(af0, bf0);
  mma(af1, bf1);

  float w00 = 0.0f;
  if (MODE == 2) w00 = Wp[0];

  // D layout: lane l, vgpr r -> m = r + (l>>4)*8, n = l&15
#pragma unroll
  for (int mt = 0; mt < MT; ++mt) {
#pragma unroll
    for (int nt = 0; nt < 4; ++nt) {
#pragma unroll
      for (int r = 0; r < 8; ++r) {
        int m = m0 + mt * 16 + lh * 8 + r;
        int n = n0 + nt * 16 + li;
        float v = acc[mt][nt][r];
        if (MODE == 0) {
          v += bias[n];
          Cb[(size_t)m * N + n] = f32_to_bf16_bits(v);
        } else if (MODE == 1) {
          v += bias[n];
          v = fmaxf(v, 0.0f);
          Hf[(size_t)m * 64 + n] = v;
          HT[(size_t)n * M + m] = f32_to_bf16_bits(v);
        } else {
          float d = deg[m];
          float outv = v * w00 / fmaxf(d, 1.0f) + Hres[(size_t)m * 64 + n];
          Of[(size_t)m * 64 + n] = (d != 0.0f) ? outv : 0.0f;
        }
      }
    }
  }
}

// ---------------------------------------------------------------- classifier
__global__ __launch_bounds__(256)
void k_clf_partial(const float* __restrict__ feat, const float* __restrict__ w,
                   float* __restrict__ partial, int len) {
  __shared__ float s0[256], s1[256];
  float a0 = 0.f, a1 = 0.f;
  for (int i = blockIdx.x * 256 + threadIdx.x; i < len; i += gridDim.x * 256) {
    float f = feat[i];
    a0 += f * w[i];
    a1 += f * w[len + i];
  }
  s0[threadIdx.x] = a0; s1[threadIdx.x] = a1;
  __syncthreads();
  for (int s = 128; s > 0; s >>= 1) {
    if (threadIdx.x < s) {
      s0[threadIdx.x] += s0[threadIdx.x + s];
      s1[threadIdx.x] += s1[threadIdx.x + s];
    }
    __syncthreads();
  }
  if (threadIdx.x == 0) { partial[blockIdx.x] = s0[0]; partial[256 + blockIdx.x] = s1[0]; }
}

__global__ __launch_bounds__(256)
void k_clf_final(const float* __restrict__ partial, const float* __restrict__ b,
                 float* __restrict__ out) {
  __shared__ float s0[256], s1[256];
  s0[threadIdx.x] = partial[threadIdx.x];
  s1[threadIdx.x] = partial[256 + threadIdx.x];
  __syncthreads();
  for (int s = 128; s > 0; s >>= 1) {
    if (threadIdx.x < s) {
      s0[threadIdx.x] += s0[threadIdx.x + s];
      s1[threadIdx.x] += s1[threadIdx.x + s];
    }
    __syncthreads();
  }
  if (threadIdx.x == 0) { out[0] = s0[0] + b[0]; out[1] = s1[0] + b[1]; }
}

// ---------------------------------------------------------------- launcher
extern "C" void kernel_launch(void* const* d_in, const int* in_sizes, int n_in,
                              void* d_out, int out_size, void* d_ws, size_t ws_size,
                              hipStream_t stream) {
  (void)in_sizes; (void)n_in; (void)out_size; (void)ws_size;
  const float* x1    = (const float*)d_in[0];
  const float* x2    = (const float*)d_in[1];
  const int*   adj1  = (const int*)d_in[2];
  const int*   adj2  = (const int*)d_in[3];
  const float* e1_w1 = (const float*)d_in[4];
  const float* e1_b1 = (const float*)d_in[5];
  const float* e1_w2 = (const float*)d_in[6];
  const float* e1_b2 = (const float*)d_in[7];
  const float* e1_w3 = (const float*)d_in[8];
  const float* e1_b3 = (const float*)d_in[9];
  const float* e2_w1 = (const float*)d_in[10];
  const float* e2_b1 = (const float*)d_in[11];
  const float* e2_w2 = (const float*)d_in[12];
  const float* e2_b2 = (const float*)d_in[13];
  const float* e2_w3 = (const float*)d_in[14];
  const float* e2_b3 = (const float*)d_in[15];
  const float* Wp    = (const float*)d_in[16];
  const float* alpha1= (const float*)d_in[17];
  const float* clf_w = (const float*)d_in[18];
  const float* clf_b = (const float*)d_in[19];
  float* out = (float*)d_out;

  // workspace bump allocator (256B aligned)
  char* p = (char*)d_ws;
  auto alloc = [&](size_t bytes) -> void* {
    void* r = (void*)p;
    p += (bytes + 255) & ~(size_t)255;
    return r;
  };
  unsigned short* xb1  = (unsigned short*)alloc((size_t)GN1 * GN1 * 2); // reused as mask1
  unsigned short* xb2  = (unsigned short*)alloc((size_t)GN2 * GN2 * 2); // reused as mask2
  unsigned short* w1b1 = (unsigned short*)alloc((size_t)256 * GN1 * 2);
  unsigned short* w1b2 = (unsigned short*)alloc((size_t)256 * GN2 * 2);
  unsigned short* w2b1 = (unsigned short*)alloc((size_t)128 * 256 * 2);
  unsigned short* w2b2 = (unsigned short*)alloc((size_t)128 * 256 * 2);
  unsigned short* w3b1 = (unsigned short*)alloc((size_t)64 * 128 * 2);
  unsigned short* w3b2 = (unsigned short*)alloc((size_t)64 * 128 * 2);
  unsigned short* c1_1 = (unsigned short*)alloc((size_t)GN1 * 256 * 2);
  unsigned short* c1_2 = (unsigned short*)alloc((size_t)GN2 * 256 * 2);
  unsigned short* c2_1 = (unsigned short*)alloc((size_t)GN1 * 128 * 2);
  unsigned short* c2_2 = (unsigned short*)alloc((size_t)GN2 * 128 * 2);
  float*          h1   = (float*)alloc((size_t)GN1 * 64 * 4);
  float*          h2   = (float*)alloc((size_t)GN2 * 64 * 4);
  unsigned short* hT1  = (unsigned short*)alloc((size_t)64 * GN1 * 2);
  unsigned short* hT2  = (unsigned short*)alloc((size_t)64 * GN2 * 2);
  float*          deg1 = (float*)alloc((size_t)GN1 * 4);
  float*          deg2 = (float*)alloc((size_t)GN2 * 4);
  float*          feat = (float*)alloc((size_t)(GN1 + GN2) * 64 * 4);
  float*          part = (float*)alloc((size_t)2 * 256 * 4);

  // 1) convert inputs / weights to bf16
  k_f32_to_bf16<<<2048, 256, 0, stream>>>(x1, xb1, (size_t)GN1 * GN1);
  k_f32_to_bf16<<<2048, 256, 0, stream>>>(x2, xb2, (size_t)GN2 * GN2);
  k_f32_to_bf16<<<512, 256, 0, stream>>>(e1_w1, w1b1, (size_t)256 * GN1);
  k_f32_to_bf16<<<512, 256, 0, stream>>>(e2_w1, w1b2, (size_t)256 * GN2);
  k_f32_to_bf16<<<64, 256, 0, stream>>>(e1_w2, w2b1, (size_t)128 * 256);
  k_f32_to_bf16<<<64, 256, 0, stream>>>(e2_w2, w2b2, (size_t)128 * 256);
  k_f32_to_bf16<<<16, 256, 0, stream>>>(e1_w3, w3b1, (size_t)64 * 128);
  k_f32_to_bf16<<<16, 256, 0, stream>>>(e2_w3, w3b2, (size_t)64 * 128);

  // 2) encoders: fc1 -> fc2 -> fc3(+relu, stores h f32 and h^T bf16)
  gemm_tn<0, 2><<<dim3(256 / 64, GN1 / 128), 128, 0, stream>>>(
      xb1, w1b1, GN1, 256, GN1, e1_b1, c1_1, nullptr, nullptr, nullptr, nullptr, nullptr, nullptr);
  gemm_tn<0, 2><<<dim3(256 / 64, GN2 / 128), 128, 0, stream>>>(
      xb2, w1b2, GN2, 256, GN2, e2_b1, c1_2, nullptr, nullptr, nullptr, nullptr, nullptr, nullptr);
  gemm_tn<0, 2><<<dim3(128 / 64, GN1 / 128), 128, 0, stream>>>(
      c1_1, w2b1, GN1, 128, 256, e1_b2, c2_1, nullptr, nullptr, nullptr, nullptr, nullptr, nullptr);
  gemm_tn<0, 2><<<dim3(128 / 64, GN2 / 128), 128, 0, stream>>>(
      c1_2, w2b2, GN2, 128, 256, e2_b2, c2_2, nullptr, nullptr, nullptr, nullptr, nullptr, nullptr);
  gemm_tn<1, 2><<<dim3(1, GN1 / 128), 128, 0, stream>>>(
      c2_1, w3b1, GN1, 64, 128, e1_b3, nullptr, h1, hT1, nullptr, nullptr, nullptr, nullptr);
  gemm_tn<1, 2><<<dim3(1, GN2 / 128), 128, 0, stream>>>(
      c2_2, w3b2, GN2, 64, 128, e2_b3, nullptr, h2, hT2, nullptr, nullptr, nullptr, nullptr);

  // 3) masked-alpha bf16 matrices + degrees (reuse xb1/xb2, dead after fc1)
  k_mask_alpha<<<GN1, 256, 0, stream>>>(adj1, alpha1, xb1, deg1, GN1);
  k_mask_alpha<<<GN2, 256, 0, stream>>>(adj2, alpha1, xb2, deg2, GN2); // alpha1[:N2,:N2], N2==N1

  // 4) attention GEMMs writing directly into concatenated feature buffer
  gemm_tn<2, 2><<<dim3(1, GN1 / 128), 128, 0, stream>>>(
      xb1, hT1, GN1, 64, GN1, nullptr, nullptr, nullptr, nullptr, Wp, deg1, h1, feat);
  gemm_tn<2, 2><<<dim3(1, GN2 / 128), 128, 0, stream>>>(
      xb2, hT2, GN2, 64, GN2, nullptr, nullptr, nullptr, nullptr, Wp, deg2, h2,
      feat + (size_t)GN1 * 64);

  // 5) classifier: [1, (N1+N2)*64] @ [(N1+N2)*64, 2] + b  (deterministic 2-stage)
  int len = (GN1 + GN2) * 64;
  k_clf_partial<<<256, 256, 0, stream>>>(feat, clf_w, part, len);
  k_clf_final<<<1, 256, 0, stream>>>(part, clf_b, out);
}
